// LinearAttention_50002009260313
// MI455X (gfx1250) — matrix operand order repaired and checked by reading, not compiled, not run on hardware
//
#include <hip/hip_runtime.h>

typedef __attribute__((ext_vector_type(16))) __bf16 v16bf;
typedef __attribute__((ext_vector_type(8)))  float  v8f;

#define ATT_B  4
#define ATT_N  4096
#define ATT_D  1024
#define ATT_H  16
#define ATT_DH 64
#define ATT_EPS 1e-6f

// ---------- helpers ----------
__device__ __forceinline__ unsigned short f2bf(float f) {
  unsigned int u = __float_as_uint(f);
  u += 0x7FFFu + ((u >> 16) & 1u);          // round-to-nearest-even
  return (unsigned short)(u >> 16);
}
__device__ __forceinline__ float bf2f(unsigned short h) {
  return __uint_as_float(((unsigned int)h) << 16);
}
__device__ __forceinline__ float elu1(float x) {   // elu(x)+1
  return x > 0.f ? x + 1.f : __expf(x);
}

union AFrag { v16bf v; unsigned int d[8]; };
union CFrag { v8f   v; float        f[8]; };

// A-matrix 16x32 bf16 fragment (ISA 7.12.2): lane m = lane&15,
// VGPR0-3: K = (lane<16?0:8)+2v..+1 ; VGPR4-7: K = (lane<16?16:24)+2(v-4)..+1
__device__ __forceinline__ v16bf load_a_frag(const unsigned short* row, int lane) {
  AFrag a;
  const unsigned int* p = (const unsigned int*)row;   // pairs of bf16
  const int hi = (lane >> 4) << 2;                    // 0 or 4 pairs
#pragma unroll
  for (int v = 0; v < 4; ++v) { a.d[v] = p[hi + v]; a.d[4 + v] = p[8 + hi + v]; }
  return a.v;
}
// B-matrix 32x16 bf16 fragment: lane n = lane&15, e<8: K = (lane<16?0:8)+e ; e>=8: K = (lane<16?16:24)+(e-8)
// (requires LDS staged as [n][k], k contiguous)
__device__ __forceinline__ v16bf load_b_frag(const unsigned short* row, int lane) {
  AFrag b;
  const unsigned int* p = (const unsigned int*)row;
  const int base = (lane >> 4) << 2;                  // 0 or 4 dwords
#pragma unroll
  for (int j = 0; j < 4; ++j) { b.d[j] = p[base + j]; b.d[4 + j] = p[8 + base + j]; }
  return b.v;
}
// low 32 bits of a generic pointer to LDS == byte offset in LDS aperture
__device__ __forceinline__ unsigned lds_off(const void* p) {
  return (unsigned)(unsigned long long)p;
}

// ---------- f32 -> bf16 convert (row-major copy) ----------
__global__ __launch_bounds__(256) void cvt_bf16_kernel(const float* __restrict__ x,
                                                       unsigned short* __restrict__ y,
                                                       long long n) {
  long long i = (long long)blockIdx.x * blockDim.x + threadIdx.x;
  if (i < n) y[i] = f2bf(x[i]);
}

// ---------- f32 (KxN) -> bf16 transposed (NxK), K=N=1024 ----------
__global__ __launch_bounds__(256) void cvt_t_kernel(const float* __restrict__ W,
                                                    unsigned short* __restrict__ Wt) {
  long long i = (long long)blockIdx.x * blockDim.x + threadIdx.x;  // 0..2^20-1
  int n = (int)(i >> 10), k = (int)(i & 1023);
  Wt[((size_t)n << 10) + k] = f2bf(W[((size_t)k << 10) + n]);
}

// ---------- 128x64-tile bf16 WMMA GEMM, double-buffered async staging ----------
// C(MxN) = op(A(MxK,row-major bf16) @ Bt(NxK,row-major bf16)^T)
// EPI: 0 = store bf16 ; 1 = elu+1 then store bf16 ; 2 = +bias, store f32
template <int EPI>
__global__ __launch_bounds__(128)
void gemm128(const unsigned short* __restrict__ A,
             const unsigned short* __restrict__ Bt,
             void* __restrict__ Cout,
             const float* __restrict__ bias,
             int M, int N, int K) {
  constexpr int PITCH = 40;                      // ushorts; 80B rows, 16B aligned
  constexpr unsigned ABUF = 128 * PITCH * 2;     // bytes per A buffer
  constexpr unsigned BBUF = 64 * PITCH * 2;      // bytes per B buffer
  __shared__ unsigned short sA[2 * 128 * PITCH]; // [par][m][k]
  __shared__ unsigned short sB[2 * 64 * PITCH];  // [par][n][k]
  const int tid  = threadIdx.x;
  const int lane = tid & 31;
  const int wave = tid >> 5;                     // 4 waves, each 32 rows x 64 cols
  const int rowBase = blockIdx.y * 128;
  const int colBase = blockIdx.x * 64;

  // staging plan: 16B chunks; thread owns k-segment s0 of rows r0+{0,32,64,96}
  // for A and rows r0+{0,32} for B.
  const int r0 = tid >> 2;                       // 0..31
  const int s0 = (tid & 3) * 8;                  // k-segment start (bf16 elems)
  const unsigned short* pA = A  + (size_t)(rowBase + r0) * K + s0;
  const unsigned short* pB = Bt + (size_t)(colBase + r0) * K + s0;
  const size_t   g32 = (size_t)32 * K;           // +32 rows in global (elems)
  const unsigned l32 = 32 * PITCH * 2;           // +32 rows in LDS (bytes)
  const unsigned lA0 = lds_off(sA) + (unsigned)(r0 * PITCH + s0) * 2;
  const unsigned lB0 = lds_off(sB) + (unsigned)(r0 * PITCH + s0) * 2;

  auto issue_tile = [&](int par, const unsigned short* ap, const unsigned short* bp) {
    unsigned la = lA0 + (unsigned)par * ABUF;
    unsigned lb = lB0 + (unsigned)par * BBUF;
    asm volatile(
        "global_load_async_to_lds_b128 %0, %4, off\n\t"
        "global_load_async_to_lds_b128 %1, %5, off\n\t"
        "global_load_async_to_lds_b128 %2, %6, off\n\t"
        "global_load_async_to_lds_b128 %3, %7, off"
        :: "v"(la), "v"(la + l32), "v"(la + 2 * l32), "v"(la + 3 * l32),
           "v"(ap), "v"(ap + g32), "v"(ap + 2 * g32), "v"(ap + 3 * g32)
        : "memory");
    asm volatile(
        "global_load_async_to_lds_b128 %0, %2, off\n\t"
        "global_load_async_to_lds_b128 %1, %3, off"
        :: "v"(lb), "v"(lb + l32), "v"(bp), "v"(bp + g32)
        : "memory");
  };

  CFrag acc[2][4];
#pragma unroll
  for (int mi = 0; mi < 2; ++mi)
#pragma unroll
    for (int t = 0; t < 4; ++t)
#pragma unroll
      for (int r = 0; r < 8; ++r) acc[mi][t].f[r] = 0.f;

  const int T = K / 32;
  issue_tile(0, pA, pB);
  for (int t = 0; t < T; ++t) {
    const int par = t & 1;
    if (t + 1 < T) {
      pA += 32;
      pB += 32;
      issue_tile((t + 1) & 1, pA, pB);
      // 6 async ops of tile t+1 may stay in flight; tile t must be complete
      asm volatile("s_wait_asynccnt 0x6" ::: "memory");
    } else {
      asm volatile("s_wait_asynccnt 0x0" ::: "memory");
    }
    __syncthreads();

    const unsigned short* bufA = sA + par * 128 * PITCH;
    const unsigned short* bufB = sB + par * 64 * PITCH;
    v16bf a0 = load_a_frag(&bufA[(wave * 32 + (lane & 15)) * PITCH], lane);
    v16bf a1 = load_a_frag(&bufA[(wave * 32 + 16 + (lane & 15)) * PITCH], lane);
#pragma unroll
    for (int tn = 0; tn < 4; ++tn) {
      v16bf bfv = load_b_frag(&bufB[(tn * 16 + (lane & 15)) * PITCH], lane);
      acc[0][tn].v = __builtin_amdgcn_wmma_f32_16x16x32_bf16(
          false, a0, false, bfv, (short)0, acc[0][tn].v, false, false);
      acc[1][tn].v = __builtin_amdgcn_wmma_f32_16x16x32_bf16(
          false, a1, false, bfv, (short)0, acc[1][tn].v, false, false);
    }
    __syncthreads();
  }

  // epilogue (C/D layout: m = r + 8*(lane>=16), n = lane&15)
#pragma unroll
  for (int mi = 0; mi < 2; ++mi)
#pragma unroll
    for (int tn = 0; tn < 4; ++tn)
#pragma unroll
      for (int r = 0; r < 8; ++r) {
        int m = wave * 32 + mi * 16 + r + 8 * (lane >> 4);
        int n = tn * 16 + (lane & 15);
        float val = acc[mi][tn].f[r];
        if (EPI == 1) val = elu1(val);
        size_t idx = (size_t)(rowBase + m) * N + colBase + n;
        if (EPI == 2)
          ((float*)Cout)[idx] = val + bias[colBase + n];
        else
          ((unsigned short*)Cout)[idx] = f2bf(val);
      }
}

// ---------- k_sum[b][c] = sum_n K[b][n][c] (8 cols per thread, b128 loads) ----------
__global__ __launch_bounds__(256) void ksum_kernel(const unsigned short* __restrict__ Kb,
                                                   float* __restrict__ ksum) {
  int idx = blockIdx.x * blockDim.x + threadIdx.x;   // 0..511
  int b = idx >> 7, c8 = (idx & 127) * 8;
  const unsigned short* p = Kb + (size_t)b * ATT_N * ATT_D + c8;
  float s[8];
#pragma unroll
  for (int j = 0; j < 8; ++j) s[j] = 0.f;
  for (int n = 0; n < ATT_N; ++n) {
    uint4 w = *(const uint4*)(p + (size_t)n * ATT_D);
    const unsigned short* e = (const unsigned short*)&w;
#pragma unroll
    for (int j = 0; j < 8; ++j) s[j] += bf2f(e[j]);
  }
#pragma unroll
  for (int j = 0; j < 8; ++j) ksum[b * ATT_D + c8 + j] = s[j];
}

// ---------- kv[b][h] = K^T V (64x64 per head) ----------
__global__ __launch_bounds__(128)
void kv_kernel(const unsigned short* __restrict__ Kb,
               const unsigned short* __restrict__ Vb,
               unsigned short* __restrict__ kvb) {
  constexpr int PITCH = 40;
  __shared__ unsigned short sK[64 * PITCH];  // [d][n]
  __shared__ unsigned short sV[64 * PITCH];  // [e][n]
  const int tid = threadIdx.x, lane = tid & 31, wave = tid >> 5;
  const int bh = blockIdx.x, b = bh >> 4, h = bh & 15;
  const size_t base = (size_t)b * ATT_N * ATT_D + h * ATT_DH;

  CFrag acc[4];
#pragma unroll
  for (int t = 0; t < 4; ++t)
#pragma unroll
    for (int r = 0; r < 8; ++r) acc[t].f[r] = 0.f;

  for (int n0 = 0; n0 < ATT_N; n0 += 32) {
    // stage transposed tiles: b128 global loads, 8x b16 LDS scatter
    for (int c = tid; c < 256; c += 128) {
      int nn = c >> 3, d8 = (c & 7) * 8;
      uint4 w = *(const uint4*)(Kb + base + (size_t)(n0 + nn) * ATT_D + d8);
      const unsigned short* e = (const unsigned short*)&w;
#pragma unroll
      for (int j = 0; j < 8; ++j) sK[(d8 + j) * PITCH + nn] = e[j];
    }
    for (int c = tid; c < 256; c += 128) {
      int nn = c >> 3, d8 = (c & 7) * 8;
      uint4 w = *(const uint4*)(Vb + base + (size_t)(n0 + nn) * ATT_D + d8);
      const unsigned short* e = (const unsigned short*)&w;
#pragma unroll
      for (int j = 0; j < 8; ++j) sV[(d8 + j) * PITCH + nn] = e[j];
    }
    __syncthreads();
    v16bf af = load_a_frag(&sK[(wave * 16 + (lane & 15)) * PITCH], lane);
#pragma unroll
    for (int t = 0; t < 4; ++t) {
      v16bf bfv = load_b_frag(&sV[(t * 16 + (lane & 15)) * PITCH], lane);
      acc[t].v = __builtin_amdgcn_wmma_f32_16x16x32_bf16(
          false, af, false, bfv, (short)0, acc[t].v, false, false);
    }
    __syncthreads();
  }
#pragma unroll
  for (int t = 0; t < 4; ++t)
#pragma unroll
    for (int r = 0; r < 8; ++r) {
      int m = wave * 16 + r + 8 * (lane >> 4);   // d
      int n = t * 16 + (lane & 15);              // e
      kvb[((size_t)bh * 64 + m) * 64 + n] = f2bf(acc[t].f[r]);
    }
}

// ---------- out = (Q @ kv) / z ; merge heads ----------
__global__ __launch_bounds__(128)
void attn_kernel(const unsigned short* __restrict__ Qb,
                 const unsigned short* __restrict__ kvb,
                 const float* __restrict__ ksum,
                 unsigned short* __restrict__ Ob) {
  constexpr int PITCH = 72;                  // 144B rows -> 16B aligned
  __shared__ unsigned short sQ[64 * PITCH];  // [row][d]
  __shared__ unsigned short sM[64 * PITCH];  // [e][d]  (= kv transposed)
  __shared__ float ksh[64];
  __shared__ float zsh[64];
  const int tid = threadIdx.x, lane = tid & 31, wave = tid >> 5;
  const int bh = blockIdx.y, b = bh >> 4, h = bh & 15;
  const int n0 = blockIdx.x * 64;
  const size_t qbase = ((size_t)b * ATT_N + n0) * ATT_D + h * ATT_DH;

  for (int c = tid; c < 512; c += 128) {     // Q tile: straight b128 copy
    int r = c >> 3, s = (c & 7) * 8;
    *(uint4*)&sQ[r * PITCH + s] =
        *(const uint4*)(Qb + qbase + (size_t)r * ATT_D + s);
  }
  for (int c = tid; c < 512; c += 128) {     // kv tile transposed into sM[e][d]
    int d = c >> 3, e8 = (c & 7) * 8;
    uint4 w = *(const uint4*)(kvb + ((size_t)bh * 64 + d) * 64 + e8);
    const unsigned short* e = (const unsigned short*)&w;
#pragma unroll
    for (int j = 0; j < 8; ++j) sM[(e8 + j) * PITCH + d] = e[j];
  }
  if (tid < 64) ksh[tid] = ksum[bh * 64 + tid];
  __syncthreads();
  if (tid < 64) {
    float s = ATT_EPS;
    for (int d = 0; d < 64; ++d) s += bf2f(sQ[tid * PITCH + d]) * ksh[d];
    zsh[tid] = s;
  }
  __syncthreads();

  CFrag acc[4];
#pragma unroll
  for (int t = 0; t < 4; ++t)
#pragma unroll
    for (int r = 0; r < 8; ++r) acc[t].f[r] = 0.f;

#pragma unroll
  for (int kc = 0; kc < 2; ++kc) {
    v16bf af = load_a_frag(&sQ[(wave * 16 + (lane & 15)) * PITCH + kc * 32], lane);
#pragma unroll
    for (int t = 0; t < 4; ++t) {
      v16bf bfv = load_b_frag(&sM[(t * 16 + (lane & 15)) * PITCH + kc * 32], lane);
      acc[t].v = __builtin_amdgcn_wmma_f32_16x16x32_bf16(
          false, af, false, bfv, (short)0, acc[t].v, false, false);
    }
  }
#pragma unroll
  for (int t = 0; t < 4; ++t)
#pragma unroll
    for (int r = 0; r < 8; ++r) {
      int m = wave * 16 + r + 8 * (lane >> 4);   // row in tile
      int n = t * 16 + (lane & 15);              // e
      Ob[((size_t)b * ATT_N + n0 + m) * ATT_D + h * ATT_DH + n] =
          f2bf(acc[t].f[r] / zsh[m]);
    }
}

// ---------- launcher ----------
extern "C" void kernel_launch(void* const* d_in, const int* in_sizes, int n_in,
                              void* d_out, int out_size, void* d_ws, size_t ws_size,
                              hipStream_t stream) {
  const float* x  = (const float*)d_in[0];
  const float* Wq = (const float*)d_in[1];
  const float* Wk = (const float*)d_in[2];
  const float* Wv = (const float*)d_in[3];
  const float* Wo = (const float*)d_in[4];
  const float* bo = (const float*)d_in[5];

  unsigned char* ws = (unsigned char*)d_ws;
  size_t off = 0;
  auto carve = [&](size_t bytes) -> void* {
    void* p = ws + off;
    off += (bytes + 255) & ~(size_t)255;
    return p;
  };
  const size_t MN = (size_t)ATT_B * ATT_N;          // 16384
  unsigned short* Xb  = (unsigned short*)carve(MN * ATT_D * 2);
  unsigned short* Wqt = (unsigned short*)carve((size_t)ATT_D * ATT_D * 2);
  unsigned short* Wkt = (unsigned short*)carve((size_t)ATT_D * ATT_D * 2);
  unsigned short* Wvt = (unsigned short*)carve((size_t)ATT_D * ATT_D * 2);
  unsigned short* Wot = (unsigned short*)carve((size_t)ATT_D * ATT_D * 2);
  unsigned short* Qb  = (unsigned short*)carve(MN * ATT_D * 2);
  unsigned short* Kb  = (unsigned short*)carve(MN * ATT_D * 2);
  unsigned short* Vb  = (unsigned short*)carve(MN * ATT_D * 2);
  unsigned short* Ab  = (unsigned short*)carve(MN * ATT_D * 2);
  unsigned short* kvb = (unsigned short*)carve((size_t)ATT_B * ATT_H * 64 * 64 * 2);
  float*          ksm = (float*)carve((size_t)ATT_B * ATT_D * 4);

  const long long nX = (long long)MN * ATT_D;       // 16,777,216
  cvt_bf16_kernel<<<(unsigned)((nX + 255) / 256), 256, 0, stream>>>(x, Xb, nX);
  cvt_t_kernel<<<4096, 256, 0, stream>>>(Wq, Wqt);
  cvt_t_kernel<<<4096, 256, 0, stream>>>(Wk, Wkt);
  cvt_t_kernel<<<4096, 256, 0, stream>>>(Wv, Wvt);
  cvt_t_kernel<<<4096, 256, 0, stream>>>(Wo, Wot);

  dim3 gProj(ATT_D / 64, (unsigned)(MN / 128));     // 16 x 128
  gemm128<1><<<gProj, 128, 0, stream>>>(Xb, Wqt, Qb, nullptr, (int)MN, ATT_D, ATT_D);
  gemm128<1><<<gProj, 128, 0, stream>>>(Xb, Wkt, Kb, nullptr, (int)MN, ATT_D, ATT_D);
  gemm128<0><<<gProj, 128, 0, stream>>>(Xb, Wvt, Vb, nullptr, (int)MN, ATT_D, ATT_D);

  ksum_kernel<<<2, 256, 0, stream>>>(Kb, ksm);
  kv_kernel<<<ATT_B * ATT_H, 128, 0, stream>>>(Kb, Vb, kvb);
  attn_kernel<<<dim3(ATT_N / 64, ATT_B * ATT_H), 128, 0, stream>>>(Qb, kvb, ksm, Ab);

  gemm128<2><<<gProj, 128, 0, stream>>>(Ab, Wot, d_out, bo, (int)MN, ATT_D, ATT_D);
}
